// SlotAttention_73160472920551
// MI455X (gfx1250) — matrix operand order repaired and checked by reading, not compiled, not run on hardware
//
#include <hip/hip_runtime.h>

// ---------------------------------------------------------------------------
// Slot Attention forward for MI455X (gfx1250), wave32 + WMMA bf16 path.
// B=32, N=4096 (64x64), Din=D=256, K=16 slots, H=4 heads (Dh=64), M=1024, 3 iters.
// GEMMs run on v_wmma_f32_16x16x32_bf16; elementwise stages in fp32.
// ---------------------------------------------------------------------------

typedef __bf16 bf16_t;
typedef __attribute__((ext_vector_type(16))) __bf16 v16bf;
typedef __attribute__((ext_vector_type(8)))  __bf16 v8bf;
typedef __attribute__((ext_vector_type(8)))  float  v8f;

#define SA_EPS   1e-8f
#define SA_LNEPS 1e-5f

static __device__ __forceinline__ bf16_t f2bf(float x) {
  unsigned u = __builtin_bit_cast(unsigned, x);
  u += 0x7FFFu + ((u >> 16) & 1u);                // round-to-nearest-even
  unsigned short h = (unsigned short)(u >> 16);
  return __builtin_bit_cast(bf16_t, h);
}

// ---- WMMA 16x16x32 bf16 fragment loads (per CDNA5 ISA layouts) -------------
// A: 16(M) x 32(K). lanes 0-15: M=lane, K chunks {0..7,16..23}; lanes 16-31:
// M=lane-16, K chunks {8..15,24..31}. Two aligned 16B loads per lane.
static __device__ __forceinline__ v16bf load_frag_a(const bf16_t* tile, int ld) {
  int lane = (int)(threadIdx.x & 31u);
  const bf16_t* p = tile + (size_t)(lane & 15) * ld + ((lane >> 4) << 3);
  v8bf lo = *(const v8bf*)p;
  v8bf hi = *(const v8bf*)(p + 16);
  v16bf f;
#pragma unroll
  for (int i = 0; i < 8; ++i) { f[i] = lo[i]; f[i + 8] = hi[i]; }
  return f;
}
// B: 32(K) x 16(N) row-major. lane: N=lane%16; K chunks {0..7,16..23} (lanes 0-15) / {8..15,24..31} (lanes 16-31).
static __device__ __forceinline__ v16bf load_frag_b(const bf16_t* tile, int ld) {
  int lane = (int)(threadIdx.x & 31u);
  const bf16_t* p = tile + (lane & 15) + (size_t)((lane >> 4) << 3) * ld;
  v16bf f;
#pragma unroll
  for (int i = 0; i < 8; ++i) { f[i] = p[(size_t)i * ld]; f[i + 8] = p[(size_t)(i + 16) * ld]; }
  return f;
}

// ---------------------------------------------------------------------------
// Generic bf16 GEMM: C[M,N] = A[M,K] * B[K,N] (+bias +residual, opt relu).
// B supplied row-major (contraction-major) = native fragment layout.
// One wave per 16x64 output strip (4 WMMA tiles sharing one A fragment).
// Requires N % 64 == 0, M % 16 == 0, K % 32 == 0.
// ---------------------------------------------------------------------------
__global__ void gemm_bf16(const bf16_t* __restrict__ A, int lda,
                          const bf16_t* __restrict__ B, int ldb,
                          const float* __restrict__ bias,
                          const float* __restrict__ residual,
                          float* __restrict__ Cf, bf16_t* __restrict__ Cb, int ldc,
                          int M, int N, int K, int relu) {
  int wave = (int)((blockIdx.x * blockDim.x + threadIdx.x) >> 5);
  int lane = (int)(threadIdx.x & 31u);
  int ng64 = N >> 6;
  int total = (M >> 4) * ng64;
  if (wave >= total) return;                       // wave-uniform: EXEC stays full
  int mt = wave / ng64, ng = wave - mt * ng64;
  const bf16_t* Ab = A + (size_t)(mt << 4) * lda;
  const bf16_t* Bb = B + (ng << 6);
  v8f c0 = {0.f, 0.f, 0.f, 0.f, 0.f, 0.f, 0.f, 0.f};
  v8f c1 = c0, c2 = c0, c3 = c0;
  for (int kk = 0; kk < K; kk += 32) {
    if (kk + 32 < K) __builtin_prefetch(Ab + kk + 32, 0, 3);   // WGP-scope prefetch
    v16bf a  = load_frag_a(Ab + kk, lda);
    const bf16_t* Bk = Bb + (size_t)kk * ldb;
    v16bf b0 = load_frag_b(Bk,      ldb);
    v16bf b1 = load_frag_b(Bk + 16, ldb);
    v16bf b2 = load_frag_b(Bk + 32, ldb);
    v16bf b3 = load_frag_b(Bk + 48, ldb);
    c0 = __builtin_amdgcn_wmma_f32_16x16x32_bf16(false, a, false, b0, (short)0, c0, false, false);
    c1 = __builtin_amdgcn_wmma_f32_16x16x32_bf16(false, a, false, b1, (short)0, c1, false, false);
    c2 = __builtin_amdgcn_wmma_f32_16x16x32_bf16(false, a, false, b2, (short)0, c2, false, false);
    c3 = __builtin_amdgcn_wmma_f32_16x16x32_bf16(false, a, false, b3, (short)0, c3, false, false);
  }
  int row0 = (mt << 4) + ((lane >> 4) << 3);
  v8f acc[4] = {c0, c1, c2, c3};
#pragma unroll
  for (int t = 0; t < 4; ++t) {
    int col = (ng << 6) + (t << 4) + (lane & 15);
    float bv = bias ? bias[col] : 0.f;
#pragma unroll
    for (int r = 0; r < 8; ++r) {
      size_t idx = (size_t)(row0 + r) * ldc + col;
      float v = acc[t][r] + bv;
      if (relu) v = fmaxf(v, 0.f);
      if (residual) v += residual[idx];
      if (Cf) Cf[idx] = v;
      if (Cb) Cb[idx] = f2bf(v);
    }
  }
}

// ---------------------------------------------------------------------------
// logits[b,k,h,n] = 0.5 * sum_d k[b,n,h*64+d] * q[b,k,h*64+d]
// A = k rows (n x d), B = qT (d x kslot). Out layout (B,K,H,N) fp32.
// ---------------------------------------------------------------------------
__global__ void logits_wmma(const bf16_t* __restrict__ kbuf,
                            const bf16_t* __restrict__ qT,
                            float* __restrict__ out) {
  int wave = (int)((blockIdx.x * blockDim.x + threadIdx.x) >> 5);
  int lane = (int)(threadIdx.x & 31u);
  if (wave >= 32 * 4 * 256) return;
  int bi  = wave >> 10;
  int rem = wave & 1023;
  int h   = rem >> 8;
  int nt  = rem & 255;
  const bf16_t* Ab = kbuf + ((size_t)(bi * 4096 + nt * 16)) * 256 + h * 64;
  const bf16_t* Bb = qT + (size_t)bi * 4096 + (size_t)(h * 64) * 16;
  v8f c = {0.f, 0.f, 0.f, 0.f, 0.f, 0.f, 0.f, 0.f};
#pragma unroll
  for (int kk = 0; kk < 64; kk += 32) {
    v16bf a = load_frag_a(Ab + kk, 256);
    v16bf b = load_frag_b(Bb + (size_t)kk * 16, 16);
    c = __builtin_amdgcn_wmma_f32_16x16x32_bf16(false, a, false, b,
                                                (short)0, c, false, false);
  }
  int ks = lane & 15;
  int n0 = nt * 16 + ((lane >> 4) << 3);
  float* dst = out + (size_t)bi * 262144 + (size_t)ks * 16384 + (size_t)h * 4096 + n0;
#pragma unroll
  for (int r = 0; r < 8; ++r) dst[r] = c[r] * 0.5f;   // SCALE = (16/4)^-0.5
}

// ---------------------------------------------------------------------------
// upd[b,k,h*64+d] = sum_n normed[b,k,h,n] * v[b,n,h*64+d]
// A = normed (kslot x n, ld=16384), B = v (n x d, ld=256). Out bf16 (B*16,256).
// ---------------------------------------------------------------------------
__global__ void upd_wmma(const bf16_t* __restrict__ normed,
                         const bf16_t* __restrict__ vbuf,
                         bf16_t* __restrict__ upd) {
  int wave = (int)((blockIdx.x * blockDim.x + threadIdx.x) >> 5);
  int lane = (int)(threadIdx.x & 31u);
  if (wave >= 512) return;
  int bi  = wave >> 4;
  int rem = wave & 15;
  int h   = rem >> 2;
  int dt  = rem & 3;
  const bf16_t* Ab = normed + (size_t)(bi * 64 + h) * 4096;          // ld 16384
  const bf16_t* Bb = vbuf + (size_t)bi * 4096 * 256 + h * 64 + dt * 16;
  v8f c = {0.f, 0.f, 0.f, 0.f, 0.f, 0.f, 0.f, 0.f};
  for (int kk = 0; kk < 4096; kk += 32) {
    v16bf a = load_frag_a(Ab + kk, 16384);
    v16bf b = load_frag_b(Bb + (size_t)kk * 256, 256);
    c = __builtin_amdgcn_wmma_f32_16x16x32_bf16(false, a, false, b,
                                                (short)0, c, false, false);
  }
  int col  = h * 64 + dt * 16 + (lane & 15);
  int row0 = bi * 16 + ((lane >> 4) << 3);
#pragma unroll
  for (int r = 0; r < 8; ++r) upd[(size_t)(row0 + r) * 256 + col] = f2bf(c[r]);
}

// ---- elementwise / reduction kernels --------------------------------------
__global__ void pack_bf16(const float* __restrict__ in, bf16_t* __restrict__ out, int n) {
  int t = (int)(blockIdx.x * blockDim.x + threadIdx.x);
  if (t < n) out[t] = f2bf(in[t]);
}

// in: rows x cols row-major -> out: cols x rows row-major (bf16)
__global__ void pack_bf16_t(const float* __restrict__ in, bf16_t* __restrict__ out,
                            int rows, int cols) {
  int t = (int)(blockIdx.x * blockDim.x + threadIdx.x);
  if (t >= rows * cols) return;
  int r = t / cols, c = t - r * cols;
  out[(size_t)c * rows + r] = f2bf(in[t]);
}

__global__ void init_slots(const float* __restrict__ noise, const float* __restrict__ mu,
                           const float* __restrict__ sigma, float* __restrict__ sf,
                           bf16_t* __restrict__ sb) {
  int t = (int)(blockIdx.x * blockDim.x + threadIdx.x);
  if (t >= 131072) return;
  int d = t & 255;
  float v = mu[d] + (fabsf(sigma[d]) + SA_EPS) * noise[t];
  sf[t] = v; sb[t] = f2bf(v);
}

// LayerNorm over width 256, one wave per row, bf16 output.
__global__ void layernorm256(const float* __restrict__ in, const float* __restrict__ g,
                             const float* __restrict__ bta, bf16_t* __restrict__ out,
                             int rows) {
  int wave = (int)((blockIdx.x * blockDim.x + threadIdx.x) >> 5);
  int lane = (int)(threadIdx.x & 31u);
  if (wave >= rows) return;
  const float* p = in + (size_t)wave * 256;
  float vals[8], s = 0.f, s2 = 0.f;
#pragma unroll
  for (int u = 0; u < 8; ++u) { float x = p[lane + u * 32]; vals[u] = x; s += x; s2 += x * x; }
#pragma unroll
  for (int m = 16; m >= 1; m >>= 1) { s += __shfl_xor(s, m, 32); s2 += __shfl_xor(s2, m, 32); }
  float mean = s * (1.f / 256.f);
  float var  = s2 * (1.f / 256.f) - mean * mean;
  float inv  = rsqrtf(var + SA_LNEPS);
  bf16_t* o = out + (size_t)wave * 256;
#pragma unroll
  for (int u = 0; u < 8; ++u) {
    int ci = lane + u * 32;
    o[ci] = f2bf((vals[u] - mean) * inv * g[ci] + bta[ci]);
  }
}

// q (B,16,256) bf16 -> qT (B,256,16) bf16
__global__ void transpose_q(const bf16_t* __restrict__ q, bf16_t* __restrict__ qT) {
  int t = (int)(blockIdx.x * blockDim.x + threadIdx.x);
  if (t >= 131072) return;
  int bi = t >> 12, rem = t & 4095, d = rem >> 4, ks = rem & 15;
  qT[t] = q[(size_t)((bi << 4) + ks) * 256 + d];
}

// 5x5 SAME conv (weight-normalized kernel), in-place per 64x64 image via LDS.
__global__ void conv5x5(float* __restrict__ buf, const float* __restrict__ ark_v,
                        const float* __restrict__ ark_g) {
  __shared__ float tile[68 * 68];
  __shared__ float w[25];
  int tid = (int)threadIdx.x;
  if (tid < 25) {
    float ss = 0.f;
#pragma unroll
    for (int i = 0; i < 25; ++i) { float x = ark_v[i]; ss += x * x; }
    w[tid] = ark_v[tid] * (ark_g[0] * rsqrtf(ss));
  }
  float* img = buf + (size_t)blockIdx.x * 4096;
  for (int i = tid; i < 68 * 68; i += 256) {
    int py = i / 68, px = i - py * 68;
    int gy = py - 2, gx = px - 2;
    tile[i] = (gy >= 0 && gy < 64 && gx >= 0 && gx < 64) ? img[gy * 64 + gx] : 0.f;
  }
  __syncthreads();
  for (int i = tid; i < 4096; i += 256) {
    int y = i >> 6, x = i & 63;
    float a = 0.f;
#pragma unroll
    for (int ky = 0; ky < 5; ++ky)
#pragma unroll
      for (int kx = 0; kx < 5; ++kx)
        a += tile[(y + ky) * 68 + (x + kx)] * w[ky * 5 + kx];
    img[i] = a;
  }
}

// softmax over K=16 (stride 16384 in (B,K,H,N) layout), in place, + EPS.
__global__ void softmax_k(float* __restrict__ buf) {
  int t = (int)(blockIdx.x * blockDim.x + threadIdx.x);
  if (t >= 32 * 4 * 4096) return;
  int bi = t >> 14, rem = t & 16383, h = rem >> 12, n = rem & 4095;
  float* p = buf + (size_t)bi * 262144 + (size_t)h * 4096 + n;
  float v[16], mx = -1e30f;
#pragma unroll
  for (int k = 0; k < 16; ++k) { v[k] = p[(size_t)k * 16384]; mx = fmaxf(mx, v[k]); }
  float s = 0.f;
#pragma unroll
  for (int k = 0; k < 16; ++k) { v[k] = __expf(v[k] - mx); s += v[k]; }
  float inv = 1.f / s;
#pragma unroll
  for (int k = 0; k < 16; ++k) p[(size_t)k * 16384] = v[k] * inv + SA_EPS;
}

// sums[g] = sum over 4096 contiguous attns values; one wave per group.
__global__ void colsum(const float* __restrict__ attns, float* __restrict__ sums) {
  int wave = (int)((blockIdx.x * blockDim.x + threadIdx.x) >> 5);
  int lane = (int)(threadIdx.x & 31u);
  if (wave >= 2048) return;
  const float* p = attns + (size_t)wave * 4096;
  float s = 0.f;
  for (int j = lane; j < 4096; j += 32) s += p[j];
#pragma unroll
  for (int m = 16; m >= 1; m >>= 1) s += __shfl_xor(s, m, 32);
  if (lane == 0) sums[wave] = s;
}

__global__ void normed_k(const float* __restrict__ attns, const float* __restrict__ sums,
                         float* __restrict__ nf, bf16_t* __restrict__ nb) {
  int t = (int)(blockIdx.x * blockDim.x + threadIdx.x);
  if (t >= 8388608) return;
  float v = attns[t] / sums[t >> 12];
  nf[t] = v; nb[t] = f2bf(v);
}

// GRU cell (torch gate order r,z,n); slots updated in place.
__global__ void gru_gate(const float* __restrict__ xg, const float* __restrict__ hg,
                         float* __restrict__ slots) {
  int t = (int)(blockIdx.x * blockDim.x + threadIdx.x);
  if (t >= 131072) return;
  int row = t >> 8, d = t & 255;
  const float* xr = xg + (size_t)row * 768;
  const float* hr = hg + (size_t)row * 768;
  float r = 1.f / (1.f + __expf(-(xr[d] + hr[d])));
  float z = 1.f / (1.f + __expf(-(xr[256 + d] + hr[256 + d])));
  float n = tanhf(xr[512 + d] + r * hr[512 + d]);
  float hp = slots[t];
  slots[t] = (1.f - z) * n + z * hp;
}

__global__ void copyf(const float* __restrict__ in, float* __restrict__ out, int n) {
  int t = (int)(blockIdx.x * blockDim.x + threadIdx.x);
  if (t < n) out[t] = in[t];
}

// ---------------------------------------------------------------------------
extern "C" void kernel_launch(void* const* d_in, const int* in_sizes, int n_in,
                              void* d_out, int out_size, void* d_ws, size_t ws_size,
                              hipStream_t stream) {
  (void)in_sizes; (void)n_in; (void)out_size; (void)ws_size;
  const float* inputs      = (const float*)d_in[0];
  const float* noise       = (const float*)d_in[1];
  const float* slots_mu    = (const float*)d_in[2];
  const float* slots_sigma = (const float*)d_in[3];
  const float* ln_in_g     = (const float*)d_in[4];
  const float* ln_in_b     = (const float*)d_in[5];
  const float* ln_slot_g   = (const float*)d_in[6];
  const float* ln_slot_b   = (const float*)d_in[7];
  const float* ln_mlp_g    = (const float*)d_in[8];
  const float* ln_mlp_b    = (const float*)d_in[9];
  const float* Wq = (const float*)d_in[10]; const float* bq = (const float*)d_in[11];
  const float* Wk = (const float*)d_in[12]; const float* bk = (const float*)d_in[13];
  const float* Wv = (const float*)d_in[14]; const float* bv = (const float*)d_in[15];
  const float* W_ih = (const float*)d_in[16]; const float* W_hh = (const float*)d_in[17];
  const float* b_ih = (const float*)d_in[18]; const float* b_hh = (const float*)d_in[19];
  const float* W1 = (const float*)d_in[20]; const float* b1 = (const float*)d_in[21];
  const float* W2 = (const float*)d_in[22]; const float* b2 = (const float*)d_in[23];
  const float* ark_v = (const float*)d_in[24];
  const float* ark_g = (const float*)d_in[25];

  float* out_slots  = (float*)d_out;                 // 131072
  float* out_attns  = out_slots + 131072;            // 8388608 (also logits workspace)
  float* out_normed = out_attns + 8388608;           // 8388608

  char* ws = (char*)d_ws;
  size_t off = 0;
  auto carve = [&](size_t bytes) -> void* {
    void* p = (void*)(ws + off);
    off += (bytes + 255) & ~(size_t)255;
    return p;
  };
  bf16_t* x_bf    = (bf16_t*)carve(33554432ull * 2);
  bf16_t* k_bf    = (bf16_t*)carve(33554432ull * 2);
  bf16_t* v_bf    = (bf16_t*)carve(33554432ull * 2);
  bf16_t* nrm_bf  = (bf16_t*)carve(8388608ull * 2);
  bf16_t* Wq_bf   = (bf16_t*)carve(65536ull * 2);
  bf16_t* Wk_bf   = (bf16_t*)carve(65536ull * 2);
  bf16_t* Wv_bf   = (bf16_t*)carve(65536ull * 2);
  bf16_t* W1_bf   = (bf16_t*)carve(262144ull * 2);
  bf16_t* W2_bf   = (bf16_t*)carve(262144ull * 2);
  bf16_t* WihT_bf = (bf16_t*)carve(196608ull * 2);
  bf16_t* WhhT_bf = (bf16_t*)carve(196608ull * 2);
  float*  slots_f = (float*)carve(131072ull * 4);
  bf16_t* slots_b = (bf16_t*)carve(131072ull * 2);
  bf16_t* ln_bf   = (bf16_t*)carve(131072ull * 2);
  bf16_t* q_bf    = (bf16_t*)carve(131072ull * 2);
  bf16_t* qT_bf   = (bf16_t*)carve(131072ull * 2);
  bf16_t* upd_bf  = (bf16_t*)carve(131072ull * 2);
  float*  xg_f    = (float*)carve(393216ull * 4);
  float*  hg_f    = (float*)carve(393216ull * 4);
  bf16_t* h1_bf   = (bf16_t*)carve(524288ull * 2);
  float*  sums_f  = (float*)carve(2048ull * 4);

  // --- one-time packs: weights -> bf16 (B-fragment native layouts) ---------
  pack_bf16<<<256, 256, 0, stream>>>(Wq, Wq_bf, 65536);
  pack_bf16<<<256, 256, 0, stream>>>(Wk, Wk_bf, 65536);
  pack_bf16<<<256, 256, 0, stream>>>(Wv, Wv_bf, 65536);
  pack_bf16<<<1024, 256, 0, stream>>>(W1, W1_bf, 262144);
  pack_bf16<<<1024, 256, 0, stream>>>(W2, W2_bf, 262144);
  pack_bf16_t<<<768, 256, 0, stream>>>(W_ih, WihT_bf, 768, 256);  // -> (256,768)
  pack_bf16_t<<<768, 256, 0, stream>>>(W_hh, WhhT_bf, 768, 256);

  init_slots<<<512, 256, 0, stream>>>(noise, slots_mu, slots_sigma, slots_f, slots_b);

  // x = LN(inputs) -> bf16; then k,v projections (WMMA, 16x64 strips)
  layernorm256<<<16384, 256, 0, stream>>>(inputs, ln_in_g, ln_in_b, x_bf, 131072);
  // waves = (131072/16)*(256/64) = 32768 -> 4096 blocks
  gemm_bf16<<<4096, 256, 0, stream>>>(x_bf, 256, Wk_bf, 256, bk, nullptr,
                                      nullptr, k_bf, 256, 131072, 256, 256, 0);
  gemm_bf16<<<4096, 256, 0, stream>>>(x_bf, 256, Wv_bf, 256, bv, nullptr,
                                      nullptr, v_bf, 256, 131072, 256, 256, 0);

  for (int it = 0; it < 3; ++it) {
    // q = LN(slots) @ Wq + bq  (waves = 32*4 = 128 -> 16 blocks)
    layernorm256<<<64, 256, 0, stream>>>(slots_f, ln_slot_g, ln_slot_b, ln_bf, 512);
    gemm_bf16<<<16, 256, 0, stream>>>(ln_bf, 256, Wq_bf, 256, bq, nullptr,
                                      nullptr, q_bf, 256, 512, 256, 256, 0);
    transpose_q<<<512, 256, 0, stream>>>(q_bf, qT_bf);

    // logits -> conv -> softmax(+eps) -> attns, all in d_out attns region
    logits_wmma<<<4096, 256, 0, stream>>>(k_bf, qT_bf, out_attns);
    conv5x5<<<2048, 256, 0, stream>>>(out_attns, ark_v, ark_g);
    softmax_k<<<2048, 256, 0, stream>>>(out_attns);

    // normed = attns / sum_n  (fp32 into d_out, bf16 A-operand for upd)
    colsum<<<256, 256, 0, stream>>>(out_attns, sums_f);
    normed_k<<<32768, 256, 0, stream>>>(out_attns, sums_f, out_normed, nrm_bf);

    // upd = normed^T @ v  (WMMA, contraction over N=4096)
    upd_wmma<<<64, 256, 0, stream>>>(nrm_bf, v_bf, upd_bf);

    // GRU: xg = upd @ W_ih^T + b_ih ; hg = slots_prev @ W_hh^T + b_hh
    // waves = 32*(768/64) = 384 -> 48 blocks
    gemm_bf16<<<48, 256, 0, stream>>>(upd_bf, 256, WihT_bf, 768, b_ih, nullptr,
                                      xg_f, nullptr, 768, 512, 768, 256, 0);
    gemm_bf16<<<48, 256, 0, stream>>>(slots_b, 256, WhhT_bf, 768, b_hh, nullptr,
                                      hg_f, nullptr, 768, 512, 768, 256, 0);
    gru_gate<<<512, 256, 0, stream>>>(xg_f, hg_f, slots_f);

    // MLP residual: slots += relu(LN(slots)@W1+b1)@W2+b2
    layernorm256<<<64, 256, 0, stream>>>(slots_f, ln_mlp_g, ln_mlp_b, ln_bf, 512);
    // waves = 32*(1024/64) = 512 -> 64 blocks
    gemm_bf16<<<64, 256, 0, stream>>>(ln_bf, 256, W1_bf, 1024, b1, nullptr,
                                      nullptr, h1_bf, 1024, 512, 1024, 256, 1);
    // waves = 32*(256/64) = 128 -> 16 blocks
    gemm_bf16<<<16, 256, 0, stream>>>(h1_bf, 1024, W2_bf, 256, b2, slots_f,
                                      slots_f, slots_b, 256, 512, 256, 1024, 0);
  }

  copyf<<<512, 256, 0, stream>>>(slots_f, out_slots, 131072);
}
